// BiLSTM_CRF_3418793968162
// MI455X (gfx1250) — compile-verified, matrix operand order checked by reading, not run on hardware
//
#include <hip/hip_runtime.h>
#include <hip/hip_bf16.h>

// BiLSTM-CRF for MI455X (gfx1250, wave32, WMMA, async LDS->mem stores).
// B=32, T=512, E=256, H=256, 4H=1024, HD=512, NT=32, START=30, STOP=31.

typedef _Float16 v16h  __attribute__((ext_vector_type(16)));
typedef _Float16 half8 __attribute__((ext_vector_type(8)));
typedef float    v8f   __attribute__((ext_vector_type(8)));
typedef int      v4i   __attribute__((ext_vector_type(4)));

#define BB   32
#define TT   512
#define EE   256
#define HH   256
#define G4H  1024
#define NTAG 32
#define START_TAG 30
#define STOP_TAG  31
#define NEGV (-10000.0f)

#if __has_builtin(__builtin_amdgcn_global_store_async_from_lds_b128) && \
    __has_builtin(__builtin_amdgcn_s_wait_asynccnt)
#define USE_ASYNC_LDS 1
// Builtin signature (from clang diagnostic): (int4 addrspace(1)*, int4 addrspace(3)*, imm, imm)
typedef __attribute__((address_space(1))) v4i v4i_g;
typedef __attribute__((address_space(3))) v4i v4i_l;
#else
#define USE_ASYNC_LDS 0
#endif

// ---------------------------------------------------------------------------
// A/B fragment loader for V_WMMA_F32_16X16X32_F16.
// A (16x32, MxK): lane l holds M=(l&15), K = kbase + (l>>4)*8 + {0..7,16..23}.
// B (32x16, KxN): mirrored — lane l holds N=(l&15) with the same K striping,
// i.e. row (nbase + (l&15)) of the row-major W for out = X @ W.T.
// ---------------------------------------------------------------------------
__device__ __forceinline__ v16h load_frag16(const _Float16* __restrict__ base,
                                            int row_base, int ld, int kbase,
                                            int lane) {
  const int r = row_base + (lane & 15);
  const int k = kbase + ((lane >> 4) << 3);
  const half8 lo = *reinterpret_cast<const half8*>(base + (size_t)r * ld + k);
  const half8 hi = *reinterpret_cast<const half8*>(base + (size_t)r * ld + k + 16);
  v16h a;
#pragma unroll
  for (int j = 0; j < 8; ++j) { a[j] = lo[j]; a[j + 8] = hi[j]; }
  return a;
}

__device__ __forceinline__ float sigf(float x) { return 1.0f / (1.0f + __expf(-x)); }

// ---------------------------------------------------------------------------
// Phase 0: fp32 -> fp16 weight conversion.
// ---------------------------------------------------------------------------
__global__ void convert_f32_to_f16(const float* __restrict__ src,
                                   _Float16* __restrict__ dst, int n) {
  int i = blockIdx.x * 256 + threadIdx.x;
  if (i < n) dst[i] = (_Float16)src[i];
}

// ---------------------------------------------------------------------------
// Phase 1: embedding gather with padding mask -> x_h[T*B][E] fp16.
// ---------------------------------------------------------------------------
__global__ void embed_kernel(const int* __restrict__ sentence,
                             const float* __restrict__ emb,
                             _Float16* __restrict__ x_h) {
  const int r = blockIdx.x;          // t*B + b
  const int e = threadIdx.x;
  const int t = r >> 5, b = r & 31;
  const int id = sentence[b * TT + t];
  const float v = (id != 0) ? emb[(size_t)id * EE + e] : 0.0f;
  x_h[(size_t)r * EE + e] = (_Float16)v;
}

// ---------------------------------------------------------------------------
// Phase 2: input GEMM  Xfrag[d] = x @ Wih[d].T + bias, stored in WMMA
// C-fragment order: Xfrag[((t*2+mi)*64 + nt)*256 + lane*8 + r] so the
// recurrent kernel can init its gate accumulators with single b128 loads.
// grid = (1024 row tiles, 8, 2 dirs), block = 256 (8 waves, wave -> N tile).
// ---------------------------------------------------------------------------
__global__ __launch_bounds__(256) void input_gemm_kernel(
    const _Float16* __restrict__ x_h,    // [16384][256]
    const _Float16* __restrict__ Wih_h,  // [2][1024][256]
    const float* __restrict__ b_f, const float* __restrict__ b_b,
    _Float16* __restrict__ Xfrag) {      // [2][16384*1024] fragment order
  const int mt   = blockIdx.x;                           // 0..1023 (row tile)
  const int dir  = blockIdx.z;
  const int lane = threadIdx.x & 31;
  const int nt   = blockIdx.y * 8 + (threadIdx.x >> 5);  // 0..63 (gate col tile)
  const _Float16* W   = Wih_h + (size_t)dir * G4H * EE;
  const float*   bias = dir ? b_b : b_f;
  v8f acc = {};
#pragma unroll
  for (int kc = 0; kc < EE / 32; ++kc) {
    v16h a = load_frag16(x_h, mt * 16, EE, kc * 32, lane);
    v16h b = load_frag16(W,   nt * 16, EE, kc * 32, lane);
    acc = __builtin_amdgcn_wmma_f32_16x16x32_f16(false, a, false, b,
                                                 (short)0, acc, false, false);
  }
  const float bn = bias[nt * 16 + (lane & 15)];  // bias folded here
  _Float16* out = Xfrag + (size_t)dir * TT * BB * G4H
                + ((size_t)mt * 64 + nt) * 256 + lane * 8;
  half8 pk;
#pragma unroll
  for (int r = 0; r < 8; ++r) pk[r] = (_Float16)(acc[r] + bn);
  *reinterpret_cast<half8*>(out) = pk;               // one b128 store
}

// ---------------------------------------------------------------------------
// Phase 3: sequential LSTM recurrence — the latency-critical path.
// One block (1024 thr = 32 waves) per direction, 512 steps.
// Wave (mi = w>>4, jt = w&15) owns cells b in [mi*16,mi*16+16),
// j in [jt*16,jt*16+16) and computes all 4 gate tiles for them:
// c stays in registers, h double-buffers through 32KB of LDS (1 barrier/step),
// gate accumulators init from fragment-ordered Xfrag (4 b128 loads),
// hs[t] streamed out via async LDS->memory b128 stores (ASYNCcnt).
// Whh (512KB fp16) stays L2-resident across all steps.
// ---------------------------------------------------------------------------
__global__ __launch_bounds__(1024) void lstm_recurrent_kernel(
    const _Float16* __restrict__ Whh_h,  // [2][1024][256]
    const _Float16* __restrict__ Xfrag,  // [2][16384*1024] fragment order
    const float* __restrict__ h0, const float* __restrict__ c0,  // [2][32][256]
    _Float16* __restrict__ hs) {         // [2][16384][256] row-major
  const int dir = blockIdx.x;
  const _Float16* Whh  = Whh_h + (size_t)dir * G4H * HH;
  const _Float16* Xf   = Xfrag + (size_t)dir * TT * BB * G4H;
  _Float16*       hs_d = hs + (size_t)dir * TT * BB * HH;

  __shared__ alignas(16) _Float16 h_lds[2][BB * HH];   // double buffer, 32KB

  const int tid  = threadIdx.x;
  const int lane = tid & 31;
  const int wave = tid >> 5;
  const int mi   = wave >> 4;
  const int jt   = wave & 15;
  const int j    = jt * 16 + (lane & 15);
  const int bofs = (lane >> 4) << 3;

  for (int idx = tid; idx < BB * HH; idx += 1024)
    h_lds[0][idx] = (_Float16)h0[dir * BB * HH + idx];

  float c_reg[8];
#pragma unroll
  for (int r = 0; r < 8; ++r) {
    const int b = mi * 16 + r + bofs;
    c_reg[r] = c0[dir * BB * HH + b * HH + j];
  }
  __syncthreads();

  for (int s = 0; s < TT; ++s) {
    const int t  = dir ? (TT - 1 - s) : s;
    const int rb = s & 1, wb = rb ^ 1;

    if (s + 1 < TT) {   // prefetch next step's fragment slice (64KB across WG)
      const int tn = dir ? (TT - 2 - s) : (s + 1);
      __builtin_prefetch(Xf + (size_t)tn * BB * G4H + tid * 32, 0, 3);
    }

    // init gate accumulators from fragment-ordered Xpre (+bias already folded)
    const _Float16* xt = Xf + ((size_t)(t * 2 + mi) * 64 + jt) * 256 + lane * 8;
    v8f acc[4];
#pragma unroll
    for (int g = 0; g < 4; ++g) {
      const half8 pk = *reinterpret_cast<const half8*>(xt + (size_t)g * 16 * 256);
#pragma unroll
      for (int r = 0; r < 8; ++r) acc[g][r] = (float)pk[r];
    }

    // gates += h @ Whh.T
#pragma unroll
    for (int kc = 0; kc < HH / 32; ++kc) {
      v16h a = load_frag16(h_lds[rb], mi * 16, HH, kc * 32, lane);
#pragma unroll
      for (int g = 0; g < 4; ++g) {
        v16h b = load_frag16(Whh, g * HH + jt * 16, HH, kc * 32, lane);
        acc[g] = __builtin_amdgcn_wmma_f32_16x16x32_f16(false, a, false, b,
                                                        (short)0, acc[g],
                                                        false, false);
      }
    }

#if USE_ASYNC_LDS
    // The async store issued 2 steps ago read h_lds[wb]; make sure it drained
    // before overwriting (ASYNCcnt is NOT covered by __syncthreads).
    __builtin_amdgcn_s_wait_asynccnt(1);
#endif

    // cell update: c in registers, new h to the write buffer
#pragma unroll
    for (int r = 0; r < 8; ++r) {
      const int b = mi * 16 + r + bofs;
      const float cn = sigf(acc[1][r]) * c_reg[r]
                     + sigf(acc[0][r]) * tanhf(acc[2][r]);
      c_reg[r] = cn;
      h_lds[wb][b * HH + j] = (_Float16)(sigf(acc[3][r]) * tanhf(cn));
    }
    __syncthreads();  // h_lds reads done (WMMA) + new h visible to all waves

    // stream hs[t] out of LDS: one b128 per thread, coalesced
    {
      _Float16* dst = hs_d + (size_t)t * BB * HH + tid * 8;
#if USE_ASYNC_LDS
      __builtin_amdgcn_global_store_async_from_lds_b128(
          (v4i_g*)dst, (v4i_l*)&h_lds[wb][tid * 8], 0, 0);
#else
      *reinterpret_cast<uint4*>(dst) =
          *reinterpret_cast<const uint4*>(&h_lds[wb][tid * 8]);
#endif
    }
  }
  // S_ENDPGM performs an implicit wait-idle, draining outstanding async stores.
}

// ---------------------------------------------------------------------------
// Phase 4: feats = concat(hs_f, hs_b) @ W_out.T + b_out  (fp16 WMMA, f32 out).
// ---------------------------------------------------------------------------
__global__ __launch_bounds__(64) void feats_gemm_kernel(
    const _Float16* __restrict__ hs,     // [2][16384][256]
    const _Float16* __restrict__ Wout_h, // [32][512]
    const float* __restrict__ b_out,     // [32]
    float* __restrict__ feats) {         // [16384][32]
  const int mt   = blockIdx.x;
  const int lane = threadIdx.x & 31;
  const int nt   = threadIdx.x >> 5;  // 0..1
  v8f acc = {};
#pragma unroll
  for (int kc = 0; kc < 16; ++kc) {
    const int kb = kc * 32;
    // k < 256 -> hs_f; k >= 256 -> hs_b (base shifted so row*256+k lands right)
    const _Float16* src = (kb < HH) ? hs : (hs + (size_t)TT * BB * HH - HH);
    v16h a = load_frag16(src, mt * 16, HH, kb, lane);
    v16h b = load_frag16(Wout_h, nt * 16, 2 * HH, kb, lane);
    acc = __builtin_amdgcn_wmma_f32_16x16x32_f16(false, a, false, b,
                                                 (short)0, acc, false, false);
  }
  const int n = nt * 16 + (lane & 15);
  const float bo = b_out[n];
#pragma unroll
  for (int r = 0; r < 8; ++r) {
    const int m = mt * 16 + r + ((lane >> 4) << 3);
    feats[(size_t)m * NTAG + n] = acc[r] + bo;
  }
}

// ---------------------------------------------------------------------------
// Phase 5: Viterbi decode + backtrace. One wave per batch element, per-batch
// 4KB transition matrix in LDS, double-buffered forward variables.
// Strict '>' ascending argmax matches jnp.argmax first-max semantics.
// ---------------------------------------------------------------------------
__global__ __launch_bounds__(32) void viterbi_kernel(
    const float* __restrict__ feats,     // [(t*32+b)][32]
    const float* __restrict__ trans,     // [32][32][32]
    unsigned char* __restrict__ bp,      // [512][32][32]
    float* __restrict__ out) {           // [32] score + [32][512] path
  const int b = blockIdx.x;
  const int j = threadIdx.x;             // next-state index
  __shared__ float tr[NTAG * NTAG];
  __shared__ float fv[2][NTAG];

  for (int idx = j; idx < NTAG * NTAG; idx += 32)
    tr[idx] = trans[(size_t)b * NTAG * NTAG + idx];
  fv[0][j] = (j == START_TAG) ? 0.0f : NEGV;
  __syncthreads();

  int p = 0;
  for (int t = 0; t < TT; ++t) {
    float best = -3.4e38f;
    int arg = 0;
    for (int prev = 0; prev < NTAG; ++prev) {
      const float s = fv[p][prev] + tr[j * NTAG + prev];
      if (s > best) { best = s; arg = prev; }
    }
    bp[(size_t)t * (BB * NTAG) + b * NTAG + j] = (unsigned char)arg;
    fv[1 - p][j] = best + feats[((size_t)t * BB + b) * NTAG + j];
    p = 1 - p;
    __syncthreads();
  }

  fv[1 - p][j] = fv[p][j] + tr[STOP_TAG * NTAG + j];   // terminal
  __syncthreads();
  if (j == 0) {
    float best = fv[1 - p][0];
    int arg = 0;
    for (int k = 1; k < NTAG; ++k) {
      const float s = fv[1 - p][k];
      if (s > best) { best = s; arg = k; }
    }
    out[b] = best;                             // path_score
    int cur = arg;
    out[BB + b * TT + (TT - 1)] = (float)cur;  // best_path[b][T-1] = last
    for (int t = TT - 1; t >= 1; --t) {
      cur = bp[(size_t)t * (BB * NTAG) + b * NTAG + cur];
      out[BB + b * TT + (t - 1)] = (float)cur; // best_path[b][t-1] = ys[t]
    }
  }
}

// ---------------------------------------------------------------------------
extern "C" void kernel_launch(void* const* d_in, const int* in_sizes, int n_in,
                              void* d_out, int out_size, void* d_ws, size_t ws_size,
                              hipStream_t stream) {
  (void)in_sizes; (void)n_in; (void)out_size; (void)ws_size;
  const int*   sentence = (const int*)d_in[0];
  const float* emb      = (const float*)d_in[1];
  const float* Wih_f    = (const float*)d_in[2];
  const float* Whh_f    = (const float*)d_in[3];
  const float* b_f      = (const float*)d_in[4];
  const float* Wih_b    = (const float*)d_in[5];
  const float* Whh_b    = (const float*)d_in[6];
  const float* b_b      = (const float*)d_in[7];
  const float* W_out    = (const float*)d_in[8];
  const float* b_out    = (const float*)d_in[9];
  const float* trans    = (const float*)d_in[10];
  const float* h0       = (const float*)d_in[11];
  const float* c0       = (const float*)d_in[12];
  float* out = (float*)d_out;

  // workspace carve-up (256B aligned)
  size_t off = 0;
  auto take = [&](size_t bytes) {
    void* p = (char*)d_ws + off;
    off += (bytes + 255) & ~(size_t)255;
    return p;
  };
  _Float16* Wih_h  = (_Float16*)take((size_t)2 * G4H * EE * 2);      // 1 MB
  _Float16* Whh_h  = (_Float16*)take((size_t)2 * G4H * HH * 2);      // 1 MB
  _Float16* Wout_h = (_Float16*)take((size_t)NTAG * 2 * HH * 2);     // 32 KB
  _Float16* x_h    = (_Float16*)take((size_t)TT * BB * EE * 2);      // 8 MB
  _Float16* Xfrag  = (_Float16*)take((size_t)2 * TT * BB * G4H * 2); // 64 MB
  _Float16* hs     = (_Float16*)take((size_t)2 * TT * BB * HH * 2);  // 16 MB
  float*    feats  = (float*)take((size_t)TT * BB * NTAG * 4);       // 2 MB
  unsigned char* bp = (unsigned char*)take((size_t)TT * BB * NTAG);  // 0.5 MB

  const int NW = G4H * EE;  // 262144 elements per weight matrix
  convert_f32_to_f16<<<(NW + 255) / 256, 256, 0, stream>>>(Wih_f, Wih_h, NW);
  convert_f32_to_f16<<<(NW + 255) / 256, 256, 0, stream>>>(Wih_b, Wih_h + NW, NW);
  convert_f32_to_f16<<<(NW + 255) / 256, 256, 0, stream>>>(Whh_f, Whh_h, NW);
  convert_f32_to_f16<<<(NW + 255) / 256, 256, 0, stream>>>(Whh_b, Whh_h + NW, NW);
  convert_f32_to_f16<<<(NTAG * 2 * HH + 255) / 256, 256, 0, stream>>>(
      W_out, Wout_h, NTAG * 2 * HH);

  embed_kernel<<<TT * BB, EE, 0, stream>>>(sentence, emb, x_h);

  input_gemm_kernel<<<dim3(TT * BB / 16, 8, 2), 256, 0, stream>>>(
      x_h, Wih_h, b_f, b_b, Xfrag);

  lstm_recurrent_kernel<<<2, 1024, 0, stream>>>(Whh_h, Xfrag, h0, c0, hs);

  feats_gemm_kernel<<<TT * BB / 16, 64, 0, stream>>>(hs, Wout_h, b_out, feats);

  viterbi_kernel<<<BB, NTAG, 0, stream>>>(feats, trans, bp, out);
}
